// ClassifyModelMOE_20220706029692
// MI455X (gfx1250) — matrix operand order repaired and checked by reading, not compiled, not run on hardware
//
#include <hip/hip_runtime.h>

typedef __bf16 bf16;
typedef __attribute__((ext_vector_type(8)))  __bf16 v8bf;
typedef __attribute__((ext_vector_type(16))) __bf16 v16bf;
typedef __attribute__((ext_vector_type(8)))  float  v8f;

#define NEXP 5
#define D_IN 3200
#define D_H  128
#define KB1  100   /* D_IN / 32 */
#define KB2  4     /* D_H  / 32 */

__device__ __forceinline__ v8f v8f_zero() {
  v8f z = {0.f,0.f,0.f,0.f,0.f,0.f,0.f,0.f};
  return z;
}

// Async copy of 32 contiguous bytes (global -> LDS), per-lane, ASYNCcnt-tracked.
// ldsOff = LDS byte address (low 32 bits of the generic shared pointer).
__device__ __forceinline__ void async_ld32(unsigned ldsOff, const bf16* g) {
  asm volatile("global_load_async_to_lds_b128 %0, %1, off"
               :: "v"(ldsOff), "v"(g) : "memory");
  asm volatile("global_load_async_to_lds_b128 %0, %1, off"
               :: "v"(ldsOff + 16u), "v"(g + 8) : "memory");
}

__device__ __forceinline__ void wait_async0() {
  asm volatile("s_wait_asynccnt 0x0" ::: "memory");
}

// ---------------------------------------------------------------------------
// Weight conversion: f32 -> bf16 pre-swizzled into the WMMA B-fragment layout.
// Layout: [e][kb][nt][lane][i], value = W[e][kb*32 + (i>=8)*16 + (lane>=16)*8 + (i&7)][nt*16 + (lane&15)]
// so in the GEMM each lane's B fragment is one contiguous, 32B-aligned load.
// ---------------------------------------------------------------------------
__global__ void cvt_w1_kernel(const float* __restrict__ w, bf16* __restrict__ out) {
  int tid  = blockIdx.x * 256 + threadIdx.x;        // NEXP*KB1*4096
  int i    = tid & 15;
  int lane = (tid >> 4) & 31;
  int nt   = (tid >> 9) & 7;
  int kb   = (tid >> 12) % KB1;
  int e    = tid / (KB1 << 12);
  int k    = kb * 32 + ((i >> 3) << 4) + ((lane >> 4) << 3) + (i & 7);
  int n    = nt * 16 + (lane & 15);
  out[tid] = (bf16)w[((size_t)e * D_IN + k) * D_H + n];
}

__global__ void cvt_w2_kernel(const float* __restrict__ w, bf16* __restrict__ out) {
  int tid  = blockIdx.x * 256 + threadIdx.x;        // NEXP*KB2*4096
  int i    = tid & 15;
  int lane = (tid >> 4) & 31;
  int nt   = (tid >> 9) & 7;
  int kb   = (tid >> 12) & 3;
  int e    = tid >> 14;
  int k    = kb * 32 + ((i >> 3) << 4) + ((lane >> 4) << 3) + (i & 7);
  int n    = nt * 16 + (lane & 15);
  out[tid] = (bf16)w[((size_t)e * D_H + k) * D_H + n];
}

// ---------------------------------------------------------------------------
// conv1 (1->16, 5x5 VALID) + ReLU + 2x2 maxpool, fused.  [B,1,28,28]->[B,16,12,12]
// ---------------------------------------------------------------------------
__global__ void conv1_pool_kernel(const float* __restrict__ x,
                                  const float* __restrict__ w,
                                  const float* __restrict__ b,
                                  float* __restrict__ pooled) {
  int idx = blockIdx.x * 256 + threadIdx.x;         // B*16*144
  int px = idx % 12, py = (idx / 12) % 12, c = (idx / 144) & 15, bb = idx / 2304;
  const float* xp = x + (size_t)bb * 784;
  const float* wp = w + c * 25;
  float bias = b[c];
  float best = -1e30f;
#pragma unroll
  for (int dy = 0; dy < 2; ++dy)
#pragma unroll
    for (int dx = 0; dx < 2; ++dx) {
      int oy = 2 * py + dy, ox = 2 * px + dx;
      float s = bias;
#pragma unroll
      for (int i = 0; i < 5; ++i)
#pragma unroll
        for (int j = 0; j < 5; ++j)
          s = fmaf(xp[(oy + i) * 28 + ox + j], wp[i * 5 + j], s);
      best = fmaxf(best, fmaxf(s, 0.f));
    }
  pooled[idx] = best;
}

// ---------------------------------------------------------------------------
// conv2 (16->32, 3x3 VALID) + ReLU, writes flattened h as bf16 [B,3200]
// ---------------------------------------------------------------------------
__global__ void conv2_kernel(const float* __restrict__ pooled,
                             const float* __restrict__ w,
                             const float* __restrict__ b,
                             bf16* __restrict__ h) {
  int idx = blockIdx.x * 256 + threadIdx.x;         // B*3200
  int px = idx % 10, py = (idx / 10) % 10, c = (idx / 100) & 31, bb = idx / 3200;
  const float* pp = pooled + (size_t)bb * 2304;
  const float* wp = w + c * 144;
  float s = b[c];
#pragma unroll
  for (int ci = 0; ci < 16; ++ci) {
    const float* prow = pp + ci * 144;
#pragma unroll
    for (int i = 0; i < 3; ++i)
#pragma unroll
      for (int j = 0; j < 3; ++j)
        s = fmaf(prow[(py + i) * 12 + px + j], wp[ci * 9 + i * 3 + j], s);
  }
  h[idx] = (bf16)fmaxf(s, 0.f);
}

// ---------------------------------------------------------------------------
// Gate: softmax(h @ gate_w + b), top-3, renormalize -> wfull [B,5]
// ---------------------------------------------------------------------------
__global__ void gate_kernel(const bf16* __restrict__ h,
                            const float* __restrict__ gw,
                            const float* __restrict__ gb,
                            float* __restrict__ wfull) {
  int lane = threadIdx.x & 31;
  int row  = blockIdx.x * 8 + (threadIdx.x >> 5);
  float acc[NEXP] = {0.f, 0.f, 0.f, 0.f, 0.f};
  const bf16* hr = h + (size_t)row * D_IN;
  for (int d = lane; d < D_IN; d += 32) {
    float hv = (float)hr[d];
#pragma unroll
    for (int e = 0; e < NEXP; ++e) acc[e] = fmaf(hv, gw[d * NEXP + e], acc[e]);
  }
#pragma unroll
  for (int e = 0; e < NEXP; ++e)
#pragma unroll
    for (int off = 16; off > 0; off >>= 1)
      acc[e] += __shfl_xor(acc[e], off, 32);
  if (lane == 0) {
    float lg[NEXP], p[NEXP];
    float mx = -1e30f;
#pragma unroll
    for (int e = 0; e < NEXP; ++e) { lg[e] = acc[e] + gb[e]; mx = fmaxf(mx, lg[e]); }
    float s = 0.f;
#pragma unroll
    for (int e = 0; e < NEXP; ++e) { p[e] = __expf(lg[e] - mx); s += p[e]; }
    float inv = 1.f / s;
#pragma unroll
    for (int e = 0; e < NEXP; ++e) p[e] *= inv;
    bool used[NEXP] = {false, false, false, false, false};
    float tsum = 0.f;
#pragma unroll
    for (int t = 0; t < 3; ++t) {
      int bi = 0; float bv = -1.f;
#pragma unroll
      for (int e = 0; e < NEXP; ++e)
        if (!used[e] && p[e] > bv) { bv = p[e]; bi = e; }
      used[bi] = true; tsum += bv;
    }
    float itn = 1.f / tsum;
#pragma unroll
    for (int e = 0; e < NEXP; ++e)
      wfull[(size_t)row * NEXP + e] = used[e] ? p[e] * itn : 0.f;
  }
}

// ---------------------------------------------------------------------------
// Expert layer 1: h1[e] = tanh(h @ W1[e] + b1[e]).
// bf16 WMMA, f32 accumulate.  Double-buffered LDS fed by async global->LDS
// copies (ASYNCcnt) so the next tile streams in while WMMAs run.
// ---------------------------------------------------------------------------
__global__ __launch_bounds__(256)
void expert1_wmma_kernel(const bf16* __restrict__ h,
                         const bf16* __restrict__ w1s,
                         const float* __restrict__ b1,
                         bf16* __restrict__ h1, int B) {
  __shared__ alignas(16) bf16 sA[2][128 * 40];   // padded stride 40 vs bank conflicts
  __shared__ alignas(32) bf16 sB[2][4096];
  int tid = threadIdx.x, lane = tid & 31, wave = tid >> 5;
  int rowBase = blockIdx.x * 128;
  int e = blockIdx.y;

  v8f acc[8];
#pragma unroll
  for (int i = 0; i < 8; ++i) acc[i] = v8f_zero();

  const int rA = tid >> 1, cA = (tid & 1) * 16;
  const bf16* gA = h + (size_t)(rowBase + rA) * D_IN + cA;
  const bf16* gB = w1s + ((size_t)e * KB1) * 4096 + tid * 16;

  unsigned offA[2], offB[2];
  offA[0] = (unsigned)(size_t)(&sA[0][0]) + (unsigned)(rA * 80 + cA * 2);
  offA[1] = (unsigned)(size_t)(&sA[1][0]) + (unsigned)(rA * 80 + cA * 2);
  offB[0] = (unsigned)(size_t)(&sB[0][0]) + (unsigned)(tid * 32);
  offB[1] = (unsigned)(size_t)(&sB[1][0]) + (unsigned)(tid * 32);

  const int rr = wave * 16 + (lane & 15);
  const int kk = (lane >> 4) * 8;

  // prologue: stage tile 0 into buffer 0
  async_ld32(offA[0], gA);
  async_ld32(offB[0], gB);

  for (int kb = 0; kb < KB1; ++kb) {
    int cur = kb & 1;
    wait_async0();          // this wave's tile-kb copies landed in LDS
    __syncthreads();        // ...and everyone else's; prev-buffer reads all done
    if (kb + 1 < KB1) {     // stream next tile into the other buffer
      async_ld32(offA[cur ^ 1], gA + (kb + 1) * 32);
      async_ld32(offB[cur ^ 1], gB + (size_t)(kb + 1) * 4096);
    }
    const bf16* a   = sA[cur];
    const bf16* bts = sB[cur];
    v8bf lo = *(const v8bf*)(a + rr * 40 + kk);
    v8bf hi = *(const v8bf*)(a + rr * 40 + kk + 16);
    v16bf af = __builtin_shufflevector(lo, hi, 0,1,2,3,4,5,6,7,8,9,10,11,12,13,14,15);
#pragma unroll
    for (int nt = 0; nt < 8; ++nt) {
      v16bf bfr = *(const v16bf*)(bts + (nt * 32 + lane) * 16);
      acc[nt] = __builtin_amdgcn_wmma_f32_16x16x32_bf16(
          false, af, false, bfr, (short)0, acc[nt], false, false);
    }
  }
  // epilogue: bias + tanh, store bf16
  int col0 = lane & 15;
  int rbase = rowBase + wave * 16 + ((lane >> 4) << 3);
#pragma unroll
  for (int nt = 0; nt < 8; ++nt) {
    int col = nt * 16 + col0;
    float bias = b1[e * D_H + col];
#pragma unroll
    for (int v = 0; v < 8; ++v) {
      float val = tanhf(acc[nt][v] + bias);
      h1[((size_t)e * B + (rbase + v)) * D_H + col] = (bf16)val;
    }
  }
}

// ---------------------------------------------------------------------------
// Expert layer 2 + top-k combine: moe[b] = sum_e wfull[b][e]*tanh(h1[e]@W2[e]+b2[e])
// Same async double-buffered pipeline over the flattened (e,kb) tile sequence.
// ---------------------------------------------------------------------------
__global__ __launch_bounds__(256)
void expert2_combine_kernel(const bf16* __restrict__ h1,
                            const bf16* __restrict__ w2s,
                            const float* __restrict__ b2,
                            const float* __restrict__ wfull,
                            float* __restrict__ moe, int B) {
  __shared__ alignas(16) bf16 sA[2][128 * 40];
  __shared__ alignas(32) bf16 sB[2][4096];
  __shared__ float sW[128 * NEXP];
  int tid = threadIdx.x, lane = tid & 31, wave = tid >> 5;
  int rowBase = blockIdx.x * 128;
  for (int i = tid; i < 128 * NEXP; i += 256)
    sW[i] = wfull[(size_t)rowBase * NEXP + i];

  v8f out[8];
#pragma unroll
  for (int i = 0; i < 8; ++i) out[i] = v8f_zero();

  const int rA = tid >> 1, cA = (tid & 1) * 16;
  const int rr = wave * 16 + (lane & 15), kk = (lane >> 4) * 8;
  const int col0 = lane & 15, mh = (lane >> 4) * 8;

  unsigned offA[2], offB[2];
  offA[0] = (unsigned)(size_t)(&sA[0][0]) + (unsigned)(rA * 80 + cA * 2);
  offA[1] = (unsigned)(size_t)(&sA[1][0]) + (unsigned)(rA * 80 + cA * 2);
  offB[0] = (unsigned)(size_t)(&sB[0][0]) + (unsigned)(tid * 32);
  offB[1] = (unsigned)(size_t)(&sB[1][0]) + (unsigned)(tid * 32);

  // prologue: tile t=0 (e=0, kb=0) into buffer 0
  async_ld32(offA[0], h1 + ((size_t)rowBase + rA) * D_H + cA);
  async_ld32(offB[0], w2s + tid * 16);

  for (int e = 0; e < NEXP; ++e) {
    v8f eacc[8];
#pragma unroll
    for (int i = 0; i < 8; ++i) eacc[i] = v8f_zero();
#pragma unroll
    for (int kb = 0; kb < KB2; ++kb) {
      int t = e * KB2 + kb, cur = t & 1;
      wait_async0();
      __syncthreads();
      if (t + 1 < NEXP * KB2) {
        int tn = t + 1, en = tn >> 2, kn = tn & 3;
        async_ld32(offA[cur ^ 1],
                   h1 + ((size_t)en * B + rowBase + rA) * D_H + kn * 32 + cA);
        async_ld32(offB[cur ^ 1], w2s + (size_t)tn * 4096 + tid * 16);
      }
      const bf16* a   = sA[cur];
      const bf16* bts = sB[cur];
      v8bf lo = *(const v8bf*)(a + rr * 40 + kk);
      v8bf hi = *(const v8bf*)(a + rr * 40 + kk + 16);
      v16bf af = __builtin_shufflevector(lo, hi, 0,1,2,3,4,5,6,7,8,9,10,11,12,13,14,15);
#pragma unroll
      for (int nt = 0; nt < 8; ++nt) {
        v16bf bfr = *(const v16bf*)(bts + (nt * 32 + lane) * 16);
        eacc[nt] = __builtin_amdgcn_wmma_f32_16x16x32_bf16(
            false, af, false, bfr, (short)0, eacc[nt], false, false);
      }
    }
#pragma unroll
    for (int nt = 0; nt < 8; ++nt) {
      int col = nt * 16 + col0;
      float bias = b2[e * D_H + col];
#pragma unroll
      for (int v = 0; v < 8; ++v) {
        int rloc = wave * 16 + mh + v;
        out[nt][v] = fmaf(sW[rloc * NEXP + e], tanhf(eacc[nt][v] + bias), out[nt][v]);
      }
    }
  }
#pragma unroll
  for (int nt = 0; nt < 8; ++nt)
#pragma unroll
    for (int v = 0; v < 8; ++v)
      moe[(size_t)(rowBase + wave * 16 + mh + v) * D_H + nt * 16 + col0] = out[nt][v];
}

// ---------------------------------------------------------------------------
// Head: softmax(moe @ sm_w + sm_b).  One wave per row.
// ---------------------------------------------------------------------------
__global__ void head_kernel(const float* __restrict__ moe,
                            const float* __restrict__ smw,
                            const float* __restrict__ smb,
                            float* __restrict__ out) {
  int lane = threadIdx.x & 31;
  int row  = blockIdx.x * 8 + (threadIdx.x >> 5);
  float acc[10] = {0.f,0.f,0.f,0.f,0.f,0.f,0.f,0.f,0.f,0.f};
#pragma unroll
  for (int t = 0; t < 4; ++t) {
    int d = lane + t * 32;
    float m = moe[(size_t)row * D_H + d];
#pragma unroll
    for (int j = 0; j < 10; ++j) acc[j] = fmaf(m, smw[d * 10 + j], acc[j]);
  }
#pragma unroll
  for (int j = 0; j < 10; ++j)
#pragma unroll
    for (int off = 16; off > 0; off >>= 1)
      acc[j] += __shfl_xor(acc[j], off, 32);
  if (lane == 0) {
    float lg[10];
    float mx = -1e30f;
#pragma unroll
    for (int j = 0; j < 10; ++j) { lg[j] = acc[j] + smb[j]; mx = fmaxf(mx, lg[j]); }
    float s = 0.f;
#pragma unroll
    for (int j = 0; j < 10; ++j) { lg[j] = __expf(lg[j] - mx); s += lg[j]; }
    float inv = 1.f / s;
#pragma unroll
    for (int j = 0; j < 10; ++j) out[(size_t)row * 10 + j] = lg[j] * inv;
  }
}

// ---------------------------------------------------------------------------
extern "C" void kernel_launch(void* const* d_in, const int* in_sizes, int n_in,
                              void* d_out, int out_size, void* d_ws, size_t ws_size,
                              hipStream_t stream) {
  const float* x   = (const float*)d_in[0];
  const float* c1w = (const float*)d_in[1];
  const float* c1b = (const float*)d_in[2];
  const float* c2w = (const float*)d_in[3];
  const float* c2b = (const float*)d_in[4];
  const float* gw  = (const float*)d_in[5];
  const float* gb  = (const float*)d_in[6];
  const float* e1w = (const float*)d_in[7];
  const float* e1b = (const float*)d_in[8];
  const float* e2w = (const float*)d_in[9];
  const float* e2b = (const float*)d_in[10];
  const float* smw = (const float*)d_in[11];
  const float* smb = (const float*)d_in[12];
  (void)n_in; (void)out_size; (void)ws_size;

  int B = in_sizes[0] / 784;   // 8192

  char* ws = (char*)d_ws;
  size_t off = 0;
  auto carve = [&](size_t bytes) -> void* {
    void* p = ws + off;
    off += (bytes + 255) & ~(size_t)255;
    return p;
  };
  float* pooled = (float*)carve((size_t)B * 2304 * sizeof(float));
  bf16*  hbf    = (bf16*) carve((size_t)B * D_IN * sizeof(bf16));
  bf16*  w1s    = (bf16*) carve((size_t)NEXP * D_IN * D_H * sizeof(bf16));
  bf16*  w2s    = (bf16*) carve((size_t)NEXP * D_H * D_H * sizeof(bf16));
  float* wfull  = (float*)carve((size_t)B * NEXP * sizeof(float));
  bf16*  h1     = (bf16*) carve((size_t)NEXP * B * D_H * sizeof(bf16));
  float* moe    = (float*)carve((size_t)B * D_H * sizeof(float));

  cvt_w1_kernel<<<(NEXP * KB1 * 4096) / 256, 256, 0, stream>>>(e1w, w1s);
  cvt_w2_kernel<<<(NEXP * KB2 * 4096) / 256, 256, 0, stream>>>(e2w, w2s);
  conv1_pool_kernel<<<(B * 16 * 144) / 256, 256, 0, stream>>>(x, c1w, c1b, pooled);
  conv2_kernel<<<(B * D_IN) / 256, 256, 0, stream>>>(pooled, c2w, c2b, hbf);
  gate_kernel<<<B / 8, 256, 0, stream>>>(hbf, gw, gb, wfull);
  expert1_wmma_kernel<<<dim3(B / 128, NEXP), 256, 0, stream>>>(hbf, w1s, e1b, h1, B);
  expert2_combine_kernel<<<B / 128, 256, 0, stream>>>(h1, w2s, e2b, wfull, moe, B);
  head_kernel<<<B / 8, 256, 0, stream>>>(moe, smw, smb, (float*)d_out);
}
